// TransformerEncoder_36636071035089
// MI455X (gfx1250) — matrix operand order repaired and checked by reading, not compiled, not run on hardware
//
#include <hip/hip_runtime.h>

typedef __attribute__((ext_vector_type(16))) __bf16 v16bf;
typedef __attribute__((ext_vector_type(8)))  __bf16 v8bf;
typedef __attribute__((ext_vector_type(8)))  float  v8f;

#define S_   512
#define B_   128
#define I_   64
#define D_   512
#define F_   2048
#define H_   8
#define L_   4
#define DK_  64
#define TOPK_ 12
#define NEG_  (-1e9f)

// ---------------------------------------------------------------------------
// WMMA + async helpers
// ---------------------------------------------------------------------------
static __device__ inline v8f wmma_bf16(v16bf a, v16bf b, v8f c) {
    return __builtin_amdgcn_wmma_f32_16x16x32_bf16(false, a, false, b, (short)0, c, false, false);
}
// async global -> LDS, 16 bytes per lane, tracked by ASYNCcnt
static __device__ inline void async_cp16(unsigned lds_addr, const void* gaddr) {
    asm volatile("global_load_async_to_lds_b128 %0, %1, off"
                 :: "v"(lds_addr), "v"(gaddr) : "memory");
}
static __device__ inline void wait_async0() {
    asm volatile("s_wait_asynccnt 0x0" ::: "memory");
}
static __device__ inline unsigned lds_off(const void* p) {
    return (unsigned)(size_t)p;     // low 32 bits of flat shared address == LDS offset
}

// A-frag 16x32 bf16 (rows K-major, stride ld): lane<16 -> row=lane, K={0..7,16..23}
//                                              lane>=16 -> row=lane-16, K={8..15,24..31}
static __device__ inline v16bf ld_a_frag(const __bf16* base, int ld, int lane) {
    int m    = lane & 15;
    int koff = (lane < 16) ? 0 : 8;
    const __bf16* p = base + m * ld + koff;
    v8bf lo = *(const v8bf*)(p);
    v8bf hi = *(const v8bf*)(p + 16);
    return __builtin_shufflevector(lo, hi, 0,1,2,3,4,5,6,7,8,9,10,11,12,13,14,15);
}
// B-frag 32x16 (KxN) from N-major LDS (tile[n*ld + k]):
// lane<16 -> col=lane, K={0..7,16..23} ; lane>=16 -> col=lane-16, K={8..15,24..31}
static __device__ inline v16bf ld_b_frag(const __bf16* base, int ld, int lane) {
    int n  = lane & 15;
    int kh = (lane < 16) ? 0 : 8;
    const __bf16* p = base + n * ld + kh;
    v8bf lo = *(const v8bf*)(p);
    v8bf hi = *(const v8bf*)(p + 16);
    return __builtin_shufflevector(lo, hi, 0,1,2,3,4,5,6,7,8,9,10,11,12,13,14,15);
}

// ---------------------------------------------------------------------------
// Generic bf16 WMMA GEMM: C(MxN) = A_bf16(MxK) @ Bt_bf16(N,K)^T (+bias)(+epilogue)
// EPI 0: store bf16                EPI 1: outF += acc + bias   (residual)
// EPI 2: store bf16(gelu(acc+b))   EPI 3: outF = (acc+b)*sqrt(D) + PE(s,n)
// 128x128 tile, 8 waves, K-step 32, async double-buffered staging
// ---------------------------------------------------------------------------
#define LDG_ 40   // padded LDS row stride (bf16): 20 words -> conflict-free b128 frags

template <int EPI>
__global__ __launch_bounds__(256) void gemm_bf16_k(
    const __bf16* __restrict__ A, const __bf16* __restrict__ Bt,
    const float* __restrict__ bias, float* outF, __bf16* outB,
    int M, int N, int K)
{
    __shared__ __align__(16) __bf16 sA[2][128 * LDG_];   // M-major
    __shared__ __align__(16) __bf16 sB[2][128 * LDG_];   // N-major

    const int n0 = blockIdx.x * 128, m0 = blockIdx.y * 128;
    const int tid = threadIdx.x, lane = tid & 31, wave = tid >> 5;
    const int wm0 = (wave & 3) * 32, wn0 = (wave >> 2) * 64;

    // 512 16B-chunks per tile; 2 chunks of A + 2 of B per thread per slab
    auto issue = [&](int buf, int kb) {
#pragma unroll
        for (int c = 0; c < 2; ++c) {
            int chunk = tid + c * 256;             // 0..511
            int row = chunk >> 2, q = (chunk & 3) * 8;
            async_cp16(lds_off(&sA[buf][row * LDG_ + q]), A  + (size_t)(m0 + row) * K + kb + q);
            async_cp16(lds_off(&sB[buf][row * LDG_ + q]), Bt + (size_t)(n0 + row) * K + kb + q);
        }
    };

    v8f acc[2][4] = {};
    const int nk = K >> 5;
    issue(0, 0);
    for (int i = 0; i < nk; ++i) {
        const int cur = i & 1;
        wait_async0();
        __syncthreads();
        if (i + 1 < nk) issue(cur ^ 1, (i + 1) << 5);
        v16bf bfr[4];
#pragma unroll
        for (int nt = 0; nt < 4; ++nt) bfr[nt] = ld_b_frag(&sB[cur][(wn0 + nt * 16) * LDG_], LDG_, lane);
#pragma unroll
        for (int mt = 0; mt < 2; ++mt) {
            v16bf afr = ld_a_frag(&sA[cur][(wm0 + mt * 16) * LDG_], LDG_, lane);
#pragma unroll
            for (int nt = 0; nt < 4; ++nt) acc[mt][nt] = wmma_bf16(afr, bfr[nt], acc[mt][nt]);
        }
        __syncthreads();
    }

    const int lane15 = lane & 15, mhi = (lane < 16) ? 0 : 8;
#pragma unroll
    for (int mt = 0; mt < 2; ++mt)
#pragma unroll
        for (int nt = 0; nt < 4; ++nt) {
            int n = n0 + wn0 + nt * 16 + lane15;
            float bv = bias ? bias[n] : 0.0f;
            float freq = 0.0f;
            if (EPI == 3) freq = __expf(-(float)(n & ~1) * (9.210340371976184f / 512.0f));
#pragma unroll
            for (int r = 0; r < 8; ++r) {
                int m = m0 + wm0 + mt * 16 + r + mhi;
                float v = acc[mt][nt][r] + bv;
                size_t off = (size_t)m * N + n;
                if (EPI == 0) outB[off] = (__bf16)v;
                else if (EPI == 1) outF[off] = outF[off] + v;
                else if (EPI == 2) {
                    float g = 0.5f * v * (1.0f + erff(v * 0.70710678118654752f));
                    outB[off] = (__bf16)g;
                } else {
                    int s = m / B_;
                    float ang = (float)s * freq;
                    float pe = (n & 1) ? __cosf(ang) : __sinf(ang);
                    outF[off] = v * 22.627416997969522f + pe;   // sqrt(512)
                }
            }
        }
}

// ---------------------------------------------------------------------------
// One-time converts: weights f32 (K,N) -> bf16 transposed (N,K);  x -> bf16 A-layout
// ---------------------------------------------------------------------------
__global__ __launch_bounds__(256) void conv_wT_k(
    const float* __restrict__ in, __bf16* __restrict__ out, int K, int N)
{
    __shared__ float tile[32][33];
    const int kb = blockIdx.x * 32, nb = blockIdx.y * 32;
    const int tx = threadIdx.x & 31, ty = threadIdx.x >> 5;
#pragma unroll
    for (int r = ty; r < 32; r += 8) tile[r][tx] = in[(size_t)(kb + r) * N + nb + tx];
    __syncthreads();
#pragma unroll
    for (int r = ty; r < 32; r += 8) out[(size_t)(nb + r) * K + kb + tx] = (__bf16)tile[tx][r];
}

__global__ __launch_bounds__(256) void conv_x_k(
    const float* __restrict__ x, __bf16* __restrict__ out)
{
    int idx = blockIdx.x * 256 + threadIdx.x;      // over S*B*I
    int i = idx & (I_ - 1), m = idx >> 6;
    int s = m / B_, b = m % B_;
    out[idx] = (__bf16)x[((size_t)b * S_ + s) * I_ + i];
}

// ---------------------------------------------------------------------------
// Batch-attention per (s,h): scores = q k^T / 8 via bf16 WMMA (exact: bf16
// products accumulate in f32), f32 top-12 mask + softmax in LDS, probs -> bf16,
// ctx = P @ V via bf16 WMMA, head-merge fused into bf16 (S,B,D) store.
// ---------------------------------------------------------------------------
#define LQK_ 72    // 64 + 8 pad (36-word stride: conflict-free)
#define LSS_ 129   // f32 scores stride
#define LPV_ 136   // 128 + 8 pad

__global__ __launch_bounds__(256) void attn_k(
    const __bf16* __restrict__ q, const __bf16* __restrict__ k,
    const __bf16* __restrict__ v, __bf16* __restrict__ ctx)
{
    __shared__ __align__(16) __bf16 sQ[128 * LQK_];      // [b][dk]
    __shared__ __align__(16) __bf16 sK[128 * LQK_];      // [b][dk] (N-major for B-frag)
    __shared__ __align__(16) __bf16 sVt[64 * LPV_];      // [dk][b] (N-major for B-frag)
    __shared__ __align__(16) __bf16 sP[128 * LPV_];      // probs [b][b]
    __shared__ float sS[128 * LSS_];                     // f32 scores

    const int sh = blockIdx.x, s = sh / H_, hh = sh % H_;
    const int tid = threadIdx.x, lane = tid & 31, wave = tid >> 5;
    const int lane15 = lane & 15, mhi = (lane < 16) ? 0 : 8;

    // stage q/k async: 512 16B-chunks each, 2 per thread per matrix
#pragma unroll
    for (int c = 0; c < 2; ++c) {
        int chunk = tid + c * 256;                 // 0..511
        int row = chunk >> 2, off = (chunk & 3) * 8;
        size_t g = ((size_t)s * B_ + row) * D_ + hh * DK_ + off;
        async_cp16(lds_off(&sQ[row * LQK_ + off]), q + g);
        async_cp16(lds_off(&sK[row * LQK_ + off]), k + g);
    }
    {   // stage V transposed (dk-major): coalesced read, strided LDS write
        int kr = tid >> 1, nh = (tid & 1) * 32;
        const __bf16* vs = v + ((size_t)s * B_ + kr) * D_ + hh * DK_ + nh;
#pragma unroll
        for (int i = 0; i < 32; ++i) sVt[(nh + i) * LPV_ + kr] = vs[i];
    }
    wait_async0();
    __syncthreads();

    // scores: wave w -> rows [16w,16w+16), 8 column tiles, K=64 (2 k-steps)
    {
        v16bf aq[2];
#pragma unroll
        for (int ks = 0; ks < 2; ++ks)
            aq[ks] = ld_a_frag(&sQ[(16 * wave) * LQK_ + ks * 32], LQK_, lane);
#pragma unroll
        for (int nt = 0; nt < 8; ++nt) {
            v8f c = {};
#pragma unroll
            for (int ks = 0; ks < 2; ++ks)
                c = wmma_bf16(aq[ks], ld_b_frag(&sK[(nt * 16) * LQK_ + ks * 32], LQK_, lane), c);
#pragma unroll
            for (int r = 0; r < 8; ++r)
                sS[(16 * wave + r + mhi) * LSS_ + 16 * nt + lane15] = c[r] * 0.125f;
        }
    }
    __syncthreads();

    // per-row top-12 threshold + masked softmax; probs stored bf16 into sP
    if (tid < 128) {
        float* row = &sS[tid * LSS_];
        __bf16* prow = &sP[tid * LPV_];
        float t[TOPK_];
#pragma unroll
        for (int i = 0; i < TOPK_; ++i) t[i] = -3.4e38f;
        for (int j = 0; j < 128; ++j) {
            float xv = row[j];
            if (xv > t[TOPK_ - 1]) {
                int p = TOPK_ - 1;
                while (p > 0 && t[p - 1] < xv) { t[p] = t[p - 1]; --p; }
                t[p] = xv;
            }
        }
        float thr = t[TOPK_ - 1], mx = t[0];
        float sum = 0.0f;
        for (int j = 0; j < 128; ++j) {
            float xv = (row[j] >= thr) ? row[j] : NEG_;
            float e = __expf(xv - mx);
            row[j] = e;
            sum += e;
        }
        float inv = 1.0f / sum;
        for (int j = 0; j < 128; ++j) prow[j] = (__bf16)(row[j] * inv);
    }
    __syncthreads();

    // ctx = P(128x128) @ V(128x64): 4 n-tiles, 4 k-steps; merged-head store
    {
        v16bf ap[4];
#pragma unroll
        for (int ks = 0; ks < 4; ++ks)
            ap[ks] = ld_a_frag(&sP[(16 * wave) * LPV_ + ks * 32], LPV_, lane);
#pragma unroll
        for (int nt = 0; nt < 4; ++nt) {
            v8f c = {};
#pragma unroll
            for (int ks = 0; ks < 4; ++ks)
                c = wmma_bf16(ap[ks], ld_b_frag(&sVt[(nt * 16) * LPV_ + ks * 32], LPV_, lane), c);
#pragma unroll
            for (int r = 0; r < 8; ++r) {
                int m = 16 * wave + r + mhi;
                ctx[((size_t)s * B_ + m) * D_ + hh * DK_ + 16 * nt + lane15] = (__bf16)c[r];
            }
        }
    }
}

// ---------------------------------------------------------------------------
// LayerNorm over D=512. FINAL=0: bf16 out. FINAL=1: f32 out, (s,b)->(b,s).
// ---------------------------------------------------------------------------
template <int FINAL>
__global__ __launch_bounds__(256) void ln_k(
    const float* __restrict__ x, const float* __restrict__ g,
    const float* __restrict__ b, __bf16* outB, float* outF)
{
    const int row = blockIdx.x, tid = threadIdx.x;
    const float* xr = x + (size_t)row * D_;
    float a0 = xr[tid], a1 = xr[tid + 256];
    __shared__ float r1[256], r2[256];
    r1[tid] = a0 + a1;
    r2[tid] = a0 * a0 + a1 * a1;
    __syncthreads();
    for (int st = 128; st > 0; st >>= 1) {
        if (tid < st) { r1[tid] += r1[tid + st]; r2[tid] += r2[tid + st]; }
        __syncthreads();
    }
    float mean = r1[0] * (1.0f / D_);
    float var  = r2[0] * (1.0f / D_) - mean * mean;
    float rstd = rsqrtf(var + 1e-5f);
    float y0 = (a0 - mean) * rstd * g[tid]       + b[tid];
    float y1 = (a1 - mean) * rstd * g[tid + 256] + b[tid + 256];
    if (FINAL == 0) {
        outB[(size_t)row * D_ + tid]       = (__bf16)y0;
        outB[(size_t)row * D_ + tid + 256] = (__bf16)y1;
    } else {
        int s = row / B_, bb = row % B_;
        size_t o = ((size_t)bb * S_ + s) * D_;
        outF[o + tid]       = y0;
        outF[o + tid + 256] = y1;
    }
}

// ---------------------------------------------------------------------------
// Launcher
// ---------------------------------------------------------------------------
extern "C" void kernel_launch(void* const* d_in, const int* in_sizes, int n_in,
                              void* d_out, int out_size, void* d_ws, size_t ws_size,
                              hipStream_t stream) {
    const float* x     = (const float*)d_in[0];
    const float* W_in  = (const float*)d_in[1];
    const float* b_in  = (const float*)d_in[2];
    const float* ln1_g = (const float*)d_in[3];
    const float* ln1_b = (const float*)d_in[4];
    const float* Wq    = (const float*)d_in[5];
    const float* Wk    = (const float*)d_in[6];
    const float* Wv    = (const float*)d_in[7];
    const float* Wo    = (const float*)d_in[8];
    const float* bo    = (const float*)d_in[9];
    const float* ln2_g = (const float*)d_in[10];
    const float* ln2_b = (const float*)d_in[11];
    const float* W1    = (const float*)d_in[12];
    const float* b1    = (const float*)d_in[13];
    const float* W2    = (const float*)d_in[14];
    const float* b2    = (const float*)d_in[15];
    const float* lnf_g = (const float*)d_in[16];
    const float* lnf_b = (const float*)d_in[17];

    const int M = S_ * B_;
    char* w = (char*)d_ws;
    float*  h    = (float*)w;  w += (size_t)M * D_ * sizeof(float);
    __bf16* ybf  = (__bf16*)w; w += (size_t)M * D_ * sizeof(__bf16);   // LN out / ctx
    __bf16* qbf  = (__bf16*)w; w += (size_t)M * D_ * sizeof(__bf16);
    __bf16* kbf  = (__bf16*)w; w += (size_t)M * D_ * sizeof(__bf16);
    __bf16* vbf  = (__bf16*)w; w += (size_t)M * D_ * sizeof(__bf16);
    __bf16* gbf  = (__bf16*)w; w += (size_t)M * F_ * sizeof(__bf16);   // gelu act
    __bf16* xbf  = (__bf16*)w; w += (size_t)M * I_ * sizeof(__bf16);
    __bf16* WinT = (__bf16*)w; w += (size_t)I_ * D_ * sizeof(__bf16);
    __bf16* WqT  = (__bf16*)w; w += (size_t)L_ * D_ * D_ * sizeof(__bf16);
    __bf16* WkT  = (__bf16*)w; w += (size_t)L_ * D_ * D_ * sizeof(__bf16);
    __bf16* WvT  = (__bf16*)w; w += (size_t)L_ * D_ * D_ * sizeof(__bf16);
    __bf16* WoT  = (__bf16*)w; w += (size_t)L_ * D_ * D_ * sizeof(__bf16);
    __bf16* W1T  = (__bf16*)w; w += (size_t)L_ * D_ * F_ * sizeof(__bf16);
    __bf16* W2T  = (__bf16*)w; w += (size_t)L_ * F_ * D_ * sizeof(__bf16);

    dim3 blk(256);
    dim3 gD(D_ / 128, M / 128);
    dim3 gF(F_ / 128, M / 128);

    // one-time converts (bf16, weights transposed to (N,K))
    conv_x_k<<<(M * I_) / 256, blk, 0, stream>>>(x, xbf);
    conv_wT_k<<<dim3(I_ / 32, D_ / 32), blk, 0, stream>>>(W_in, WinT, I_, D_);
    for (int l = 0; l < L_; ++l) {
        size_t oD = (size_t)l * D_ * D_, oF = (size_t)l * D_ * F_;
        conv_wT_k<<<dim3(D_ / 32, D_ / 32), blk, 0, stream>>>(Wq + oD, WqT + oD, D_, D_);
        conv_wT_k<<<dim3(D_ / 32, D_ / 32), blk, 0, stream>>>(Wk + oD, WkT + oD, D_, D_);
        conv_wT_k<<<dim3(D_ / 32, D_ / 32), blk, 0, stream>>>(Wv + oD, WvT + oD, D_, D_);
        conv_wT_k<<<dim3(D_ / 32, D_ / 32), blk, 0, stream>>>(Wo + oD, WoT + oD, D_, D_);
        conv_wT_k<<<dim3(D_ / 32, F_ / 32), blk, 0, stream>>>(W1 + oF, W1T + oF, D_, F_);
        conv_wT_k<<<dim3(F_ / 32, D_ / 32), blk, 0, stream>>>(W2 + oF, W2T + oF, F_, D_);
    }

    // input projection: h = (x^T @ W_in + b_in)*sqrt(D) + PE  (EPI 3)
    gemm_bf16_k<3><<<gD, blk, 0, stream>>>(xbf, WinT, b_in, h, nullptr, M, D_, I_);

    for (int l = 0; l < L_; ++l) {
        size_t oD = (size_t)l * D_ * D_, oF = (size_t)l * D_ * F_;
        ln_k<0><<<M, blk, 0, stream>>>(h, ln1_g + l * D_, ln1_b + l * D_, ybf, nullptr);
        gemm_bf16_k<0><<<gD, blk, 0, stream>>>(ybf, WqT + oD, nullptr, nullptr, qbf, M, D_, D_);
        gemm_bf16_k<0><<<gD, blk, 0, stream>>>(ybf, WkT + oD, nullptr, nullptr, kbf, M, D_, D_);
        gemm_bf16_k<0><<<gD, blk, 0, stream>>>(ybf, WvT + oD, nullptr, nullptr, vbf, M, D_, D_);
        attn_k<<<S_ * H_, blk, 0, stream>>>(qbf, kbf, vbf, ybf);
        gemm_bf16_k<1><<<gD, blk, 0, stream>>>(ybf, WoT + oD, bo + l * D_, h, nullptr, M, D_, D_);
        ln_k<0><<<M, blk, 0, stream>>>(h, ln2_g + l * D_, ln2_b + l * D_, ybf, nullptr);
        gemm_bf16_k<2><<<gF, blk, 0, stream>>>(ybf, W1T + oF, b1 + l * F_, nullptr, gbf, M, F_, D_);
        gemm_bf16_k<1><<<gD, blk, 0, stream>>>(gbf, W2T + oF, b2 + l * D_, h, nullptr, M, D_, F_);
    }
    ln_k<1><<<M, blk, 0, stream>>>(h, lnf_g, lnf_b, nullptr, (float*)d_out);
}
